// dgl_simpgcn_18047452578204
// MI455X (gfx1250) — matrix operand order repaired and checked by reading, not compiled, not run on hardware
//
#include <hip/hip_runtime.h>
#include <hip/hip_bf16.h>
#include <stdint.h>

#define FDIM 512
#define GAMMA_C 0.1f

typedef __attribute__((ext_vector_type(16))) __bf16 v16bf;
typedef __attribute__((ext_vector_type(8)))  float  v8f;
typedef __attribute__((ext_vector_type(4)))  unsigned int v4u;
typedef __attribute__((ext_vector_type(8)))  int v8i_t;
typedef __attribute__((ext_vector_type(4)))  int v4i_t;

__device__ __forceinline__ unsigned short f32_bf16(float f) {
  union { float f; unsigned int u; } c; c.f = f;
  unsigned int u = c.u;
  u += 0x7fffu + ((u >> 16) & 1u);   // round-to-nearest-even
  return (unsigned short)(u >> 16);
}

// ---------------- zero fill ----------------
__global__ void zero_f4(float4* p, long long n4) {
  long long i = (long long)blockIdx.x * blockDim.x + threadIdx.x;
  if (i < n4) p[i] = make_float4(0.f, 0.f, 0.f, 0.f);
}

// ---------------- fp32 -> bf16 ----------------
__global__ void cvt_bf16_k(const float* __restrict__ in, unsigned short* __restrict__ out, int n) {
  int i = blockIdx.x * blockDim.x + threadIdx.x;
  if (i < n) out[i] = f32_bf16(in[i]);
}

// ------------- gates: s = sigmoid(H.sw+sb), Kd = H.kw+kb (one wave / node) -------------
__global__ void gates_k(const float* __restrict__ H,
                        const float* __restrict__ sw, const float* __restrict__ sbp,
                        const float* __restrict__ kw, const float* __restrict__ kbp,
                        float* __restrict__ s, float* __restrict__ Kd, int n) {
  int node = (blockIdx.x * blockDim.x + threadIdx.x) >> 5;
  int lane = threadIdx.x & 31;
  if (node >= n) return;
  const float* h = H + (size_t)node * FDIM;
  float as = 0.f, ak = 0.f;
#pragma unroll
  for (int t = 0; t < FDIM / 32; ++t) {
    int f = lane + 32 * t;
    float hv = h[f];
    as += hv * sw[f];
    ak += hv * kw[f];
  }
#pragma unroll
  for (int m = 16; m >= 1; m >>= 1) {
    as += __shfl_xor(as, m, 32);
    ak += __shfl_xor(ak, m, 32);
  }
  if (lane == 0) {
    float sv = as + sbp[0];
    s[node]  = 1.f / (1.f + __expf(-sv));
    Kd[node] = ak + kbp[0];
  }
}

// ------------- COO SpMM scatter: out[row,:] += val * H[col,:] (one wave / nnz) -------------
__global__ void spmm_k(const int* __restrict__ rows, const int* __restrict__ cols,
                       const float* __restrict__ vals, const float* __restrict__ H,
                       float* __restrict__ out, int nnz) {
  int e = (blockIdx.x * blockDim.x + threadIdx.x) >> 5;
  int lane = threadIdx.x & 31;
  if (e >= nnz) return;
  int r = rows[e], c = cols[e];
  float v = vals[e];
  const float* hc = H + (size_t)c * FDIM;
  float* orow = out + (size_t)r * FDIM;
#pragma unroll
  for (int t = 0; t < FDIM / 32; ++t) {
    int f = lane + 32 * t;
    unsafeAtomicAdd(&orow[f], v * hc[f]);   // global_atomic_add_f32
  }
}

// ------------- PH = s*AH + (1-s)*SfH + gamma*Kd*H, cast bf16 -------------
__global__ void combine_k(const float* __restrict__ s, const float* __restrict__ Kd,
                          const float* __restrict__ AH, const float* __restrict__ SfH,
                          const float* __restrict__ H, unsigned short* __restrict__ PHb,
                          long long total) {
  long long i = (long long)blockIdx.x * blockDim.x + threadIdx.x;
  if (i >= total) return;
  int n = (int)(i >> 9);    // /512
  float sv = s[n];
  float ph = sv * AH[i] + (1.f - sv) * SfH[i] + GAMMA_C * (Kd[n] * H[i]);
  PHb[i] = f32_bf16(ph);
}

// ------------- WMMA GEMM with TDM-staged weight panel -------------
// Block: 256 threads (8 waves), 128 rows x 64 cols of output.
// Wave w owns rows m0 = blk.x*128 + w*16 and 4 adjacent 16x16 col tiles.
// Weight panel [512 x 64] bf16 is DMA'd to LDS by the Tensor Data Mover, then
// transposed once into a padded [col][K] layout for ds_load_b128 fragments.
template <int FOUT, bool RELU>
__global__ __launch_bounds__(256) void wmma_gemm_k(
    const unsigned short* __restrict__ Abf,   // [N,512] bf16
    const unsigned short* __restrict__ Wbf,   // [512,FOUT] bf16 row-major
    const float* __restrict__ bias,           // [FOUT]
    float* __restrict__ out,                  // [N,FOUT] fp32
    int n) {
  constexpr int CB  = 64;                     // column block (4 wmma tiles)
  constexpr int LDB = FDIM + 8;               // padded K stride (bank-conflict dodge)
  extern __shared__ unsigned short smem[];
  unsigned short* Braw = smem;                // [FDIM][CB]  (TDM destination, 64KB)
  unsigned short* Bst  = smem + FDIM * CB;    // [CB][LDB]   (transposed, ~65KB)

  int j0 = blockIdx.y * CB;

  if (threadIdx.x < 32) {
    // ---- Tensor DMA descriptor (CDNA5 ISA sec.8): 2D tile CB x FDIM ----
    unsigned long long gaddr = (unsigned long long)(uintptr_t)(Wbf + j0);
    unsigned lds = (unsigned)(uintptr_t)Braw;
    v4u g0 = { 1u,                                        // count=1 (valid user D#)
               lds,                                       // lds_addr
               (unsigned)(gaddr & 0xffffffffu),           // global_addr[31:0]
               (unsigned)((gaddr >> 32) & 0x1ffffffu)     // global_addr[56:32]
                 | (2u << 30) };                          // type=2 ("image")
    unsigned td0 = (unsigned)FOUT;                        // tensor_dim0 (row length)
    unsigned td1 = (unsigned)FDIM;                        // tensor_dim1 (#rows)
    unsigned long long st0 = (unsigned long long)FOUT;    // dim0 stride (elements)
    v8i_t g1 = { (int)(1u << 16),                               // data_size=1 (2B), mask=0
                 (int)((td0 & 0xffffu) << 16),                  // tensor_dim0[15:0]
                 (int)((td0 >> 16) | ((td1 & 0xffffu) << 16)),  // dim0[31:16] | dim1[15:0]
                 (int)((td1 >> 16) | ((unsigned)CB << 16)),     // dim1[31:16] | tile_dim0
                 (int)FDIM,                                     // tile_dim1 | tile_dim2=0
                 (int)(st0 & 0xffffffffu),                      // stride0[31:0]
                 (int)((st0 >> 32) & 0xffffu),                  // stride0[47:32] | stride1=0
                 0 };
    v4i_t gz4 = { 0, 0, 0, 0 };
    v8i_t gz8 = { 0, 0, 0, 0, 0, 0, 0, 0 };
    __builtin_amdgcn_tensor_load_to_lds(g0, g1, gz4, gz4, gz8, 0);  // tensor_load_to_lds
    __builtin_amdgcn_s_wait_tensorcnt(0);                           // s_wait_tensorcnt 0
  }
  __syncthreads();

  // ---- transpose [k][n] -> padded [n][k] (keeps ds_load_b128 fragment reads) ----
  for (int idx = threadIdx.x; idx < FDIM * CB; idx += 256) {
    int k = idx >> 6, nn = idx & 63;
    Bst[nn * LDB + k] = Braw[idx];
  }
  __syncthreads();

  int wave = threadIdx.x >> 5, lane = threadIdx.x & 31;
  int g = lane >> 4;                          // lane half (ISA 16-bit fragment layout)
  int m0 = blockIdx.x * 128 + wave * 16;
  int arow = m0 + (lane & 15);
  if (arow >= n) arow = n - 1;                // clamp: EXEC must stay all-ones for WMMA
  const __bf16* A     = ((const __bf16*)Abf) + (size_t)arow * FDIM + 16 * g;
  const __bf16* Bbase = ((const __bf16*)Bst) + 16 * g;

  v8f accs[4] = {};
#pragma unroll
  for (int ks = 0; ks < FDIM; ks += 32) {
    v16bf a;
#pragma unroll
    for (int j = 0; j < 8; ++j) {             // A 16x32: lane half g holds K {16g..16g+15}
      a[j]     = A[ks + j];
      a[j + 8] = A[ks + j + 8];
    }
#pragma unroll
    for (int t = 0; t < 4; ++t) {             // reuse A fragment across 4 column tiles
      const __bf16* Bl = Bbase + (size_t)(t * 16 + (lane & 15)) * LDB;
      v16bf b;
#pragma unroll
      for (int j = 0; j < 16; ++j)            // B 32x16: lane half g holds K {16g..16g+15}
        b[j] = Bl[ks + j];
      accs[t] = __builtin_amdgcn_wmma_f32_16x16x32_bf16(false, a, false, b, (short)0,
                                                        accs[t], false, false);
    }
  }

#pragma unroll
  for (int t = 0; t < 4; ++t) {
    int col = j0 + t * 16 + (lane & 15);
    float bv = bias[col];
#pragma unroll
    for (int rr = 0; rr < 8; ++rr) {          // C/D: lanes 0-15 rows 0..7, lanes 16-31 rows 8..15
      int orow = m0 + 8 * g + rr;
      if (orow < n) {
        float v = accs[t][rr] + bv;
        if (RELU) v = fmaxf(v, 0.f);
        out[(size_t)orow * FOUT + col] = v;
      }
    }
  }
}

extern "C" void kernel_launch(void* const* d_in, const int* in_sizes, int n_in,
                              void* d_out, int out_size, void* d_ws, size_t ws_size,
                              hipStream_t stream) {
  const float* features = (const float*)d_in[0];
  const int*   ar  = (const int*)d_in[1];
  const int*   acx = (const int*)d_in[2];
  const float* av  = (const float*)d_in[3];
  const int*   sr  = (const int*)d_in[4];
  const int*   scx = (const int*)d_in[5];
  const float* svv = (const float*)d_in[6];
  const float* W0  = (const float*)d_in[7];
  const float* W0b = (const float*)d_in[8];
  const float* W1  = (const float*)d_in[9];
  const float* W1b = (const float*)d_in[10];
  const float* s0w = (const float*)d_in[11];
  const float* s0b = (const float*)d_in[12];
  const float* s1w = (const float*)d_in[13];
  const float* s1b = (const float*)d_in[14];
  const float* k0w = (const float*)d_in[15];
  const float* k0b = (const float*)d_in[16];
  const float* k1w = (const float*)d_in[17];
  const float* k1b = (const float*)d_in[18];

  const int n     = in_sizes[0] / FDIM;   // 100000
  const int nnz_a = in_sizes[1];          // E + N
  const int nnz_s = in_sizes[4];          // 2*N*K
  const long long totNF = (long long)n * FDIM;

  // ---- workspace carve ----
  char* ws = (char*)d_ws;
  size_t off = 0;
  auto carve = [&](size_t bytes) {
    void* p = ws + off;
    off += (bytes + 255) & ~(size_t)255;
    return p;
  };
  float* sgate = (float*)carve((size_t)n * 4);
  float* kdiag = (float*)carve((size_t)n * 4);
  float* AH    = (float*)carve((size_t)totNF * 4);
  float* SfH   = (float*)carve((size_t)totNF * 4);
  unsigned short* PHb  = (unsigned short*)carve((size_t)totNF * 2);
  float* H1    = (float*)carve((size_t)totNF * 4);
  unsigned short* W0bf = (unsigned short*)carve((size_t)FDIM * FDIM * 2);
  unsigned short* W1bf = (unsigned short*)carve((size_t)FDIM * 64 * 2);

  // ---- weight conversion (bf16 for WMMA) ----
  cvt_bf16_k<<<(FDIM * FDIM + 255) / 256, 256, 0, stream>>>(W0, W0bf, FDIM * FDIM);
  cvt_bf16_k<<<(FDIM * 64 + 255) / 256, 256, 0, stream>>>(W1, W1bf, FDIM * 64);

  const long long n4 = totNF / 4;
  const int zb = (int)((n4 + 255) / 256);
  const int gatesBlocks = (n * 32 + 255) / 256;
  const int spmmA = (int)(((long long)nnz_a * 32 + 255) / 256);
  const int spmmS = (int)(((long long)nnz_s * 32 + 255) / 256);
  const int combBlocks = (int)((totNF + 255) / 256);
  const int mTiles = (n + 127) / 128;
  const size_t gemmSmem = (size_t)(FDIM * 64 + 64 * (FDIM + 8)) * 2;  // ~130KB < 320KB WGP LDS

  // ================= layer 1 (features -> H1, ReLU) =================
  zero_f4<<<zb, 256, 0, stream>>>((float4*)AH, n4);
  zero_f4<<<zb, 256, 0, stream>>>((float4*)SfH, n4);
  gates_k<<<gatesBlocks, 256, 0, stream>>>(features, s0w, s0b, k0w, k0b, sgate, kdiag, n);
  spmm_k<<<spmmA, 256, 0, stream>>>(ar, acx, av, features, AH, nnz_a);
  spmm_k<<<spmmS, 256, 0, stream>>>(sr, scx, svv, features, SfH, nnz_s);
  combine_k<<<combBlocks, 256, 0, stream>>>(sgate, kdiag, AH, SfH, features, PHb, totNF);
  {
    dim3 grid(mTiles, FDIM / 64);
    wmma_gemm_k<FDIM, true><<<grid, 256, gemmSmem, stream>>>(PHb, W0bf, W0b, H1, n);
  }

  // ================= layer 2 (H1 -> out) =================
  zero_f4<<<zb, 256, 0, stream>>>((float4*)AH, n4);
  zero_f4<<<zb, 256, 0, stream>>>((float4*)SfH, n4);
  gates_k<<<gatesBlocks, 256, 0, stream>>>(H1, s1w, s1b, k1w, k1b, sgate, kdiag, n);
  spmm_k<<<spmmA, 256, 0, stream>>>(ar, acx, av, H1, AH, nnz_a);
  spmm_k<<<spmmS, 256, 0, stream>>>(sr, scx, svv, H1, SfH, nnz_s);
  combine_k<<<combBlocks, 256, 0, stream>>>(sgate, kdiag, AH, SfH, H1, PHb, totNF);
  {
    dim3 grid(mTiles, 64 / 64);
    wmma_gemm_k<64, false><<<grid, 256, gemmSmem, stream>>>(PHb, W1bf, W1b, (float*)d_out, n);
  }
}
